// VMamba_Decoder1_34626026340769
// MI455X (gfx1250) — matrix operand order repaired and checked by reading, not compiled, not run on hardware
//
#include <hip/hip_runtime.h>
#include <math.h>

#define EPS  1e-5f
#define Bn   2
#define Ll   4096
#define DM   64
#define DI   128
#define Ns   16
#define Rr   4
#define Kd   4
#define Mrows (Bn*Ll)
#define LPB  8          // scan positions per k_xproj block

typedef __attribute__((ext_vector_type(16))) _Float16 v16h;
typedef __attribute__((ext_vector_type(8)))  float    v8f;
typedef unsigned int v4u __attribute__((ext_vector_type(4)));
typedef int          v8i __attribute__((ext_vector_type(8)));
typedef int          v4i __attribute__((ext_vector_type(4)));

#if defined(__has_builtin)
#if __has_builtin(__builtin_amdgcn_tensor_load_to_lds)
#define USE_TDM 1
#endif
#endif

__device__ __forceinline__ float silu_f(float x){ return x / (1.f + __expf(-x)); }
__device__ __forceinline__ float softplus_f(float x){ return (x > 20.f) ? x : log1pf(__expf(x)); }
__device__ __forceinline__ float gelu_f(float x){ return 0.5f * x * (1.f + erff(x * 0.70710678118654752f)); }

// position l in scan-direction k  ->  spatial index into xconv (row-major h*64+w)
__device__ __forceinline__ int scan_index(int k, int l){
  switch (k & 3){
    case 0:  return l;
    case 1:  return ((l & 63) << 6) | (l >> 6);             // column-major
    case 2:  return Ll - 1 - l;                             // reversed row-major
    default: { int t = Ll - 1 - l; return ((t & 63) << 6) | (t >> 6); }
  }
}

// ---------------- K0: reduce_channel + patch_embed + LN -> feat (output) and x ----------------
__global__ void k_embed(const float* __restrict__ enc, const float* __restrict__ rc_w,
                        const float* __restrict__ pe_w, const float* __restrict__ pe_b,
                        const float* __restrict__ ln_g, const float* __restrict__ ln_b,
                        float* __restrict__ feat_out, float* __restrict__ x_out){
  __shared__ float s_enc[2*DM];
  __shared__ float s_feat[DM];
  __shared__ float s_y[DM];
  __shared__ float s_red[DM];
  int t = threadIdx.x;                       // 128 threads
  int b = blockIdx.x / Ll, l = blockIdx.x % Ll;
  s_enc[t] = enc[(b*2*DM + t)*Ll + l];
  __syncthreads();
  if (t < DM){
    const float* wr = rc_w + t*2*DM;
    float f = 0.f;
#pragma unroll 8
    for (int c = 0; c < 2*DM; ++c) f += wr[c]*s_enc[c];
    feat_out[(b*DM + t)*Ll + l] = f;
    s_feat[t] = f;
  }
  __syncthreads();
  if (t < DM){
    const float* wr = pe_w + t*DM;
    float y = pe_b[t];
#pragma unroll 8
    for (int c = 0; c < DM; ++c) y += wr[c]*s_feat[c];
    s_y[t] = y; s_red[t] = y;
  }
  __syncthreads();
  for (int s = DM/2; s > 0; s >>= 1){ if (t < s) s_red[t] += s_red[t+s]; __syncthreads(); }
  float mean = s_red[0] * (1.f/DM);
  __syncthreads();
  if (t < DM){ float d = s_y[t]-mean; s_red[t] = d*d; }
  __syncthreads();
  for (int s = DM/2; s > 0; s >>= 1){ if (t < s) s_red[t] += s_red[t+s]; __syncthreads(); }
  float rstd = rsqrtf(s_red[0]*(1.f/DM) + EPS);
  if (t < DM) x_out[(b*Ll + l)*DM + t] = (s_y[t]-mean)*rstd*ln_g[t] + ln_b[t];
}

// ---------------- LayerNorm over 64 channels, one row per wave ----------------
__global__ void k_ln64(const float* __restrict__ x, const float* __restrict__ g,
                       const float* __restrict__ bta, float* __restrict__ y){
  int lane = threadIdx.x & 31, wave = threadIdx.x >> 5;
  int row = blockIdx.x*8 + wave;
  float v0 = x[row*64 + lane], v1 = x[row*64 + 32 + lane];
  float s = v0+v1;
#pragma unroll
  for (int o=16;o;o>>=1) s += __shfl_xor(s,o,32);
  float mean = s*(1.f/64.f);
  float d0=v0-mean, d1=v1-mean, q = d0*d0+d1*d1;
#pragma unroll
  for (int o=16;o;o>>=1) q += __shfl_xor(q,o,32);
  float r = rsqrtf(q*(1.f/64.f)+EPS);
  y[row*64+lane]      = d0*r*g[lane]      + bta[lane];
  y[row*64+32+lane]   = d1*r*g[32+lane]   + bta[32+lane];
}

// -------- WMMA GEMM: Y[M,O] = act(X[M,C]*W[O,C]^T + bias), 32x64 tile per wave --------
// 2 A-fragments x 4 B-fragments -> 8 v_wmma per K-step; gfx1250 wave32 layouts (CDNA5 ISA)
template<int ACT, int ACCUM>
__global__ void k_gemm_wmma(const float* __restrict__ X, const float* __restrict__ Wm,
                            const float* __restrict__ bias, float* __restrict__ Y,
                            int C, int O){
  int lane = threadIdx.x;                  // 32 lanes, EXEC all ones
  int to = blockIdx.x * 64, tm = blockIdx.y * 32;
  int hsel = lane >> 4;                    // 0/1: wave half
  int l15  = lane & 15;
  v8f acc[8] = {};
  const float* xr0 = X  + (tm + l15)*C;          // A rows m..m+15
  const float* xr1 = xr0 + 16*C;                 // A rows m+16..m+31
  const float* wr0 = Wm + (to + l15)*C;          // B cols (weights row-major (O,C))
  for (int k0 = 0; k0 < C; k0 += 32){
    v16h a0, a1, b0, b1, b2, b3;
    const float* p0 = xr0 + k0 + hsel*16;        // A: elem i -> k0 + hsel*16 + i
    const float* p1 = xr1 + k0 + hsel*16;
#pragma unroll
    for (int i=0;i<8;i++){
      a0[i]   = (_Float16)p0[i];  a0[8+i] = (_Float16)p0[8+i];
      a1[i]   = (_Float16)p1[i];  a1[8+i] = (_Float16)p1[8+i];
    }
    const float* q = wr0 + k0 + hsel*16;         // B: elem i -> k0 + hsel*16 + i
#pragma unroll
    for (int i=0;i<16;i++){
      b0[i] = (_Float16)q[i];
      b1[i] = (_Float16)q[16*C + i];
      b2[i] = (_Float16)q[32*C + i];
      b3[i] = (_Float16)q[48*C + i];
    }
    acc[0] = __builtin_amdgcn_wmma_f32_16x16x32_f16(false, a0, false, b0, (short)0, acc[0], false, false);
    acc[1] = __builtin_amdgcn_wmma_f32_16x16x32_f16(false, a0, false, b1, (short)0, acc[1], false, false);
    acc[2] = __builtin_amdgcn_wmma_f32_16x16x32_f16(false, a0, false, b2, (short)0, acc[2], false, false);
    acc[3] = __builtin_amdgcn_wmma_f32_16x16x32_f16(false, a0, false, b3, (short)0, acc[3], false, false);
    acc[4] = __builtin_amdgcn_wmma_f32_16x16x32_f16(false, a1, false, b0, (short)0, acc[4], false, false);
    acc[5] = __builtin_amdgcn_wmma_f32_16x16x32_f16(false, a1, false, b1, (short)0, acc[5], false, false);
    acc[6] = __builtin_amdgcn_wmma_f32_16x16x32_f16(false, a1, false, b2, (short)0, acc[6], false, false);
    acc[7] = __builtin_amdgcn_wmma_f32_16x16x32_f16(false, a1, false, b3, (short)0, acc[7], false, false);
  }
#pragma unroll
  for (int j=0;j<4;j++){
    int col = to + j*16 + l15;
    float bv = bias ? bias[col] : 0.f;
#pragma unroll
    for (int i=0;i<2;i++){
      int mb = tm + i*16 + hsel*8;               // D: vgpr r -> row r (+8 upper half)
#pragma unroll
      for (int r=0;r<8;r++){
        float v = acc[i*4+j][r] + bv;
        if (ACT==1) v = gelu_f(v);
        if (ACCUM) Y[(mb+r)*O + col] += v;
        else       Y[(mb+r)*O + col]  = v;
      }
    }
  }
}

// ---------------- depthwise 3x3 conv + SiLU on xc (first 128 chans of xz) ----------------
__global__ void k_dwconv(const float* __restrict__ xz, const float* __restrict__ cw,
                         const float* __restrict__ cb, float* __restrict__ xconv){
  int d = blockIdx.x % DI, b = blockIdx.x / DI;
  float wv[9];
#pragma unroll
  for (int i=0;i<9;i++) wv[i] = cw[d*9 + i];
  float bias = cb[d];
  for (int l = threadIdx.x; l < Ll; l += blockDim.x){
    int h = l >> 6, wc = l & 63;
    float acc = bias;
#pragma unroll
    for (int kh=0;kh<3;kh++){
      int hh = h + kh - 1; if ((unsigned)hh >= 64u) continue;
#pragma unroll
      for (int kw=0;kw<3;kw++){
        int wwv = wc + kw - 1; if ((unsigned)wwv >= 64u) continue;
        acc += wv[kh*3+kw] * xz[(b*Ll + (hh<<6) + wwv)*256 + d];
      }
    }
    xconv[(b*DI + d)*Ll + l] = silu_f(acc);
  }
}

// -------- x-proj (36x128) + dt-proj + softplus; weights staged in LDS by the TDM --------
// LDS row stride 129 dwords (TDM pad: +1 dword every 128) -> conflict-free dot reads.
__global__ void k_xproj(const float* __restrict__ xconv, const float* __restrict__ xpw,
                        const float* __restrict__ dtw, const float* __restrict__ dtb,
                        float* __restrict__ dt_out, float* __restrict__ Bs_out,
                        float* __restrict__ Cs_out){
  __shared__ float s_w[36*129];
  __shared__ float s_x[DI];
  __shared__ float s_p[36];
  int t = threadIdx.x;                       // 128 threads
  int grp = blockIdx.x;                      // (b,k,l-group)
  int l0 = (grp % (Ll/LPB)) * LPB;
  int k  = (grp / (Ll/LPB)) % Kd;
  int b  =  grp / ((Ll/LPB) * Kd);

#ifdef USE_TDM
  if ((threadIdx.x >> 5) == 0){              // one wave issues the DMA
    unsigned lds_off = (unsigned)(size_t)(void*)s_w;
    unsigned long long ga = (unsigned long long)(size_t)(xpw + k*(36*DI));
    // D# group0: count=1 | lds_addr | global_addr | type=2
    v4u g0 = { 1u, lds_off, (unsigned)(ga & 0xffffffffu),
               (unsigned)((ga >> 32) & 0x01ffffffu) | 0x80000000u };
    // D# group1: data_size=4B, pad_enable, pad_interval=128dw, pad_amount=1dw,
    //            tensor_dim0=tile_dim0=stride=4608, tensor_dim1=tile_dim1=1
    v8i g1 = { (int)0x01920000, (int)0x12000000, (int)0x00010000, (int)0x12000000,
               1, 4608, 0, 0 };
    v4i gz = { 0, 0, 0, 0 };
#if defined(__clang_major__) && (__clang_major__ >= 23)
    v8i gz8 = { 0,0,0,0,0,0,0,0 };
    __builtin_amdgcn_tensor_load_to_lds(g0, g1, gz, gz, gz8, 0);
#else
    __builtin_amdgcn_tensor_load_to_lds(g0, g1, gz, gz, 0);
#endif
    __builtin_amdgcn_s_wait_tensorcnt(0);
  }
#else
  for (int i = t; i < 36*DI; i += blockDim.x)
    s_w[(i/DI)*129 + (i%DI)] = xpw[k*(36*DI) + i];
#endif
  // dt-proj weights for this thread's channel (hoisted)
  float wdt[Rr];
#pragma unroll
  for (int r=0;r<Rr;r++) wdt[r] = dtw[(k*DI + t)*Rr + r];
  float bdt = dtb[k*DI + t];
  __syncthreads();

  for (int li = 0; li < LPB; ++li){
    int l = l0 + li;
    s_x[t] = xconv[(b*DI + t)*Ll + scan_index(k, l)];
    __syncthreads();
    if (t < 36){
      const float* wr = s_w + t*129;
      float acc = 0.f;
#pragma unroll 8
      for (int d=0; d<DI; ++d) acc += wr[d]*s_x[d];
      s_p[t] = acc;
    }
    __syncthreads();
    int base = (b*Kd + k)*Ll + l;
    float v = bdt;
#pragma unroll
    for (int r=0;r<Rr;r++) v += wdt[r]*s_p[r];
    dt_out[base*DI + t] = softplus_f(v);
    if (t < Ns)        Bs_out[base*Ns + t]      = s_p[Rr + t];
    else if (t < 2*Ns) Cs_out[base*Ns + (t-Ns)] = s_p[Rr + t];
    __syncthreads();                         // protect s_x/s_p for next l
  }
}

// ---------------- selective scan: 1024 independent chains, 16 states in regs ----------------
__global__ void k_scan(const float* __restrict__ xconv, const float* __restrict__ dt_in,
                       const float* __restrict__ Bs, const float* __restrict__ Cs,
                       const float* __restrict__ A_logs, const float* __restrict__ Ds,
                       float* __restrict__ ys){
  int idx = blockIdx.x*blockDim.x + threadIdx.x;  // 0..1023
  int d = idx % DI;
  int k = (idx / DI) % Kd;
  int b = idx / (DI*Kd);
  float Ac[Ns];
#pragma unroll
  for (int n=0;n<Ns;n++) Ac[n] = -__expf(A_logs[(k*DI + d)*Ns + n]);
  float Dc = Ds[k*DI + d];
  float h[Ns];
#pragma unroll
  for (int n=0;n<Ns;n++) h[n]=0.f;
  const float* xc = xconv + (b*DI + d)*Ll;
  int base0 = (b*Kd + k)*Ll;
  for (int l=0; l<Ll; ++l){
    int base = base0 + l;
    float dt = dt_in[base*DI + d];
    float u  = xc[scan_index(k,l)];
    float du = dt*u;
    float y = 0.f;
    const float* Bp = Bs + base*Ns;
    const float* Cp = Cs + base*Ns;
#pragma unroll
    for (int n=0;n<Ns;n++){
      float a = __expf(dt*Ac[n]);
      h[n] = a*h[n] + du*Bp[n];
      y += h[n]*Cp[n];
    }
    ys[base*DI + d] = y + Dc*u;
    if ((l & 7) == 0 && l + 8 < Ll)
      __builtin_prefetch(dt_in + (base+8)*DI + d, 0, 1);   // global_prefetch_b8
  }
}

// ---------------- cross-merge + out-LN + SiLU(z) gate ----------------
__global__ void k_merge(const float* __restrict__ ys, const float* __restrict__ og,
                        const float* __restrict__ ob, const float* __restrict__ xz,
                        float* __restrict__ gout){
  __shared__ float s_red[DI];
  int t = threadIdx.x;                     // 128 threads
  int l = blockIdx.x % Ll, b = blockIdx.x / Ll;
  int h = l >> 6, w = l & 63;
  int idx1 = (w << 6) | h;
  int b4 = b*Kd;
  float y = ys[((b4+0)*Ll + l)*DI + t]
          + ys[((b4+2)*Ll + (Ll-1-l))*DI + t]
          + ys[((b4+1)*Ll + idx1)*DI + t]
          + ys[((b4+3)*Ll + (Ll-1-idx1))*DI + t];
  s_red[t] = y; __syncthreads();
  for (int s=DI/2;s>0;s>>=1){ if (t<s) s_red[t]+=s_red[t+s]; __syncthreads(); }
  float mean = s_red[0]*(1.f/DI);
  __syncthreads();
  float dv = y-mean; s_red[t]=dv*dv; __syncthreads();
  for (int s=DI/2;s>0;s>>=1){ if (t<s) s_red[t]+=s_red[t+s]; __syncthreads(); }
  float r = rsqrtf(s_red[0]*(1.f/DI)+EPS);
  float z = xz[(b*Ll + l)*256 + DI + t];
  gout[(b*Ll + l)*DI + t] = (dv*r*og[t] + ob[t]) * silu_f(z);
}

// ---------------- head conv1: 64->32, 3x3, leaky relu ----------------
__global__ void k_head1(const float* __restrict__ x, const float* __restrict__ w1,
                        float* __restrict__ h1){
  __shared__ float s_w[DM*9];
  int o = blockIdx.x % 32, b = blockIdx.x / 32;
  for (int i = threadIdx.x; i < DM*9; i += blockDim.x) s_w[i] = w1[o*DM*9 + i];
  __syncthreads();
  for (int l = threadIdx.x; l < Ll; l += blockDim.x){
    int h = l >> 6, w = l & 63;
    float acc = 0.f;
#pragma unroll
    for (int kh=0;kh<3;kh++){
      int hh = h+kh-1; if ((unsigned)hh >= 64u) continue;
#pragma unroll
      for (int kw=0;kw<3;kw++){
        int wwv = w+kw-1; if ((unsigned)wwv >= 64u) continue;
        const float* xr = x + (b*Ll + (hh<<6)+wwv)*DM;
        int tap = kh*3+kw;
#pragma unroll 8
        for (int c=0;c<DM;c++) acc += xr[c]*s_w[c*9 + tap];
      }
    }
    h1[(b*32+o)*Ll + l] = acc > 0.f ? acc : 0.01f*acc;
  }
}

// ---------------- head conv2: 32->1, 3x3, + inp, sigmoid ----------------
__global__ void k_head2(const float* __restrict__ h1, const float* __restrict__ w2,
                        const float* __restrict__ inp, float* __restrict__ out){
  __shared__ float s_w[32*9];
  int b = blockIdx.x;
  for (int i = threadIdx.x; i < 32*9; i += blockDim.x) s_w[i] = w2[i];
  __syncthreads();
  for (int l = threadIdx.x; l < Ll; l += blockDim.x){
    int h = l >> 6, w = l & 63;
    float acc = inp[b*Ll + l];
    for (int c=0;c<32;c++){
      const float* hr = h1 + (b*32+c)*Ll;
#pragma unroll
      for (int kh=0;kh<3;kh++){
        int hh=h+kh-1; if((unsigned)hh>=64u) continue;
#pragma unroll
        for (int kw=0;kw<3;kw++){
          int wwv=w+kw-1; if((unsigned)wwv>=64u) continue;
          acc += s_w[c*9+kh*3+kw]*hr[(hh<<6)+wwv];
        }
      }
    }
    out[b*Ll + l] = 1.f/(1.f+__expf(-acc));
  }
}

extern "C" void kernel_launch(void* const* d_in, const int* in_sizes, int n_in,
                              void* d_out, int out_size, void* d_ws, size_t ws_size,
                              hipStream_t stream){
  (void)in_sizes; (void)n_in; (void)out_size; (void)ws_size;
  const float* inp      = (const float*)d_in[0];
  const float* enc      = (const float*)d_in[1];
  const float* rc_w     = (const float*)d_in[2];
  const float* pe_w     = (const float*)d_in[3];
  const float* pe_b     = (const float*)d_in[4];
  const float* pe_ln_g  = (const float*)d_in[5];
  const float* pe_ln_b  = (const float*)d_in[6];
  const float* n1_g     = (const float*)d_in[7];
  const float* n1_b     = (const float*)d_in[8];
  const float* inproj_w = (const float*)d_in[9];
  const float* conv_w   = (const float*)d_in[10];
  const float* conv_b   = (const float*)d_in[11];
  const float* xproj_w  = (const float*)d_in[12];
  const float* dtproj_w = (const float*)d_in[13];
  const float* dtproj_b = (const float*)d_in[14];
  const float* A_logs   = (const float*)d_in[15];
  const float* Ds       = (const float*)d_in[16];
  const float* onorm_g  = (const float*)d_in[17];
  const float* onorm_b  = (const float*)d_in[18];
  const float* outproj_w= (const float*)d_in[19];
  const float* n2_g     = (const float*)d_in[20];
  const float* n2_b     = (const float*)d_in[21];
  const float* fc1_w    = (const float*)d_in[22];
  const float* fc1_b    = (const float*)d_in[23];
  const float* fc2_w    = (const float*)d_in[24];
  const float* fc2_b    = (const float*)d_in[25];
  const float* head1_w  = (const float*)d_in[26];
  const float* head2_w  = (const float*)d_in[27];

  float* out  = (float*)d_out;             // out: [0, 8192)
  float* feat = out + Bn*Ll;               // feat: [8192, +524288)

  float* ws = (float*)d_ws;  size_t off = 0;
  float* x     = ws + off;  off += (size_t)Mrows*DM;        // 2 MB
  float* tmpA  = ws + off;  off += (size_t)Mrows*256;       // 8 MB (xz, later fc1 out)
  float* tmpB  = ws + off;  off += (size_t)Mrows*DM;        // 2 MB (LN rows)
  float* xconv = ws + off;  off += (size_t)Bn*DI*Ll;        // 4 MB
  float* dtb   = ws + off;  off += (size_t)Bn*Kd*Ll*DI;     // 16 MB
  float* Bsb   = ws + off;  off += (size_t)Bn*Kd*Ll*Ns;     // 2 MB
  float* Csb   = ws + off;  off += (size_t)Bn*Kd*Ll*Ns;     // 2 MB
  float* ysb   = ws + off;  off += (size_t)Bn*Kd*Ll*DI;     // 16 MB
  float* gbuf  = ws + off;  off += (size_t)Mrows*DI;        // 4 MB
  float* h1b   = ws + off;  off += (size_t)Bn*32*Ll;        // 1 MB

  k_embed<<<Bn*Ll, 128, 0, stream>>>(enc, rc_w, pe_w, pe_b, pe_ln_g, pe_ln_b, feat, x);

  for (int blk = 0; blk < 4; ++blk){
    k_ln64<<<Mrows/8, 256, 0, stream>>>(x, n1_g + blk*DM, n1_b + blk*DM, tmpB);
    k_gemm_wmma<0,0><<<dim3(256/64, Mrows/32), 32, 0, stream>>>(
        tmpB, inproj_w + blk*256*DM, nullptr, tmpA, DM, 256);
    k_dwconv<<<Bn*DI, 256, 0, stream>>>(tmpA, conv_w + blk*DI*9, conv_b + blk*DI, xconv);
    k_xproj<<<Bn*Kd*(Ll/LPB), 128, 0, stream>>>(xconv, xproj_w + blk*Kd*36*DI,
        dtproj_w + blk*Kd*DI*Rr, dtproj_b + blk*Kd*DI, dtb, Bsb, Csb);
    k_scan<<<4, 256, 0, stream>>>(xconv, dtb, Bsb, Csb,
        A_logs + blk*Kd*DI*Ns, Ds + blk*Kd*DI, ysb);
    k_merge<<<Bn*Ll, DI, 0, stream>>>(ysb, onorm_g + blk*DI, onorm_b + blk*DI, tmpA, gbuf);
    k_gemm_wmma<0,1><<<dim3(DM/64, Mrows/32), 32, 0, stream>>>(
        gbuf, outproj_w + blk*DM*DI, nullptr, x, DI, DM);            // residual add
    k_ln64<<<Mrows/8, 256, 0, stream>>>(x, n2_g + blk*DM, n2_b + blk*DM, tmpB);
    k_gemm_wmma<1,0><<<dim3(256/64, Mrows/32), 32, 0, stream>>>(
        tmpB, fc1_w + blk*256*DM, fc1_b + blk*256, tmpA, DM, 256);   // GELU
    k_gemm_wmma<0,1><<<dim3(DM/64, Mrows/32), 32, 0, stream>>>(
        tmpA, fc2_w + blk*DM*256, fc2_b + blk*DM, x, 256, DM);       // residual add
  }
  k_head1<<<Bn*32, 256, 0, stream>>>(x, head1_w, h1b);
  k_head2<<<Bn, 256, 0, stream>>>(h1b, head2_w, inp, out);
}
